// _ScaledDotProductAttention_19112604467820
// MI455X (gfx1250) — matrix operand order repaired and checked by reading, not compiled, not run on hardware
//
#include <hip/hip_runtime.h>

// ---------------------------------------------------------------------------
// Scaled dot-product attention for MI455X (gfx1250), flash-attention style.
//   q: [32, 8, 512, 64] f32      k: [32, 8, 64, 512] f32 (pre-transposed)
//   v: [32, 8, 512, 64] f32      mask: [8, 512, 512] bool (True -> -inf)
//   out: [32, 8, 512, 64] f32
// bf16 WMMA (v_wmma_f32_16x16x32_bf16), f32 accumulate, exp2-domain softmax,
// software-pipelined K/V staging (free truncating bf16 stores via d16_hi),
// v_permlane16 row reductions.
// ---------------------------------------------------------------------------

typedef __attribute__((ext_vector_type(8)))  float  v8f;
typedef __attribute__((ext_vector_type(8)))  __bf16 v8bf;
typedef __attribute__((ext_vector_type(16))) __bf16 v16bf;

#define N_G   32      // B*C
#define N_H   8
#define N_L   512
#define N_D   64
#define QTILE 128     // query rows per block (8 waves x 16 rows)
#define KTILE 64      // kv rows per inner iteration
#define NJ    (N_L / KTILE)
#define NEG_BIG (-1.0e30f)
// 1/sqrt(64) * log2(e): scores land directly in exp2 domain
#define QSCALE 0.18033688011112042f

// fp32 -> bf16 by truncation: just the float's top 16 bits. When stored to a
// 16-bit LDS slot the compiler emits ds_store_b16_d16_hi of the raw fp32 reg.
__device__ __forceinline__ unsigned short bftrunc(float f) {
    return (unsigned short)(__float_as_uint(f) >> 16);
}
// Packed pair conversion (used for Q staging, which does vector stores).
__device__ __forceinline__ unsigned int pack2(float lo, float hi) {
#if __has_builtin(__builtin_amdgcn_cvt_pk_bf16_f32)
    typedef __attribute__((ext_vector_type(2))) __bf16 v2bf;
    v2bf r = __builtin_amdgcn_cvt_pk_bf16_f32(lo, hi);
    return __builtin_bit_cast(unsigned int, r);
#elif __has_builtin(__builtin_amdgcn_perm)
    // v_perm_b32: one op, truncating pack {hi[31:16], lo[31:16]}
    return __builtin_amdgcn_perm(__float_as_uint(hi), __float_as_uint(lo),
                                 0x07060302u);
#else
    return (__float_as_uint(lo) >> 16) | (__float_as_uint(hi) & 0xFFFF0000u);
#endif
}
__device__ __forceinline__ float fast_exp2(float x) {
#if __has_builtin(__builtin_amdgcn_exp2f)
    return __builtin_amdgcn_exp2f(x);
#else
    return exp2f(x);
#endif
}
__device__ __forceinline__ v16bf cat16(v8bf lo, v8bf hi) {
    return __builtin_shufflevector(lo, hi, 0,1,2,3,4,5,6,7,8,9,10,11,12,13,14,15);
}

// XOR-k lane swap within each 16-lane group: v_permlane16_b32 (pure VALU,
// no LDS pipeline round trip like ds_bpermute).
__device__ __forceinline__ float xor16(float x, int k, unsigned s0, unsigned s1) {
#if __has_builtin(__builtin_amdgcn_permlane16)
    unsigned u = __float_as_uint(x);
    u = __builtin_amdgcn_permlane16(u, u, s0, s1, false, false);
    return __uint_as_float(u);
#else
    return __shfl_xor(x, k, 32);
#endif
}
__device__ __forceinline__ float redmax16(float x) {
    x = fmaxf(x, xor16(x, 1, 0x67452301u, 0xEFCDAB89u));
    x = fmaxf(x, xor16(x, 2, 0x54761032u, 0xDCFE98BAu));
    x = fmaxf(x, xor16(x, 4, 0x32107654u, 0xBA98FEDCu));
    x = fmaxf(x, xor16(x, 8, 0xFEDCBA98u, 0x76543210u));
    return x;
}
__device__ __forceinline__ float redsum16(float x) {
    x += xor16(x, 1, 0x67452301u, 0xEFCDAB89u);
    x += xor16(x, 2, 0x54761032u, 0xDCFE98BAu);
    x += xor16(x, 4, 0x32107654u, 0xBA98FEDCu);
    x += xor16(x, 8, 0xFEDCBA98u, 0x76543210u);
    return x;
}

__global__ __launch_bounds__(256)
void sdpa_flash_bf16_kernel(const float* __restrict__ q,
                            const float* __restrict__ k,
                            const float* __restrict__ v,
                            const unsigned char* __restrict__ mask,
                            float* __restrict__ out) {
    // LDS staging (bf16 as ushort). 48 KB total.
    __shared__ alignas(16) unsigned short sQ[QTILE * N_D];   // [row][d]
    __shared__ alignas(16) unsigned short sK[KTILE * N_D];   // [m][d]  (transposed K slab)
    __shared__ alignas(16) unsigned short sV[N_D * KTILE];   // [d][m]  (transposed V slab)
    __shared__ alignas(16) unsigned short sP[8 * 16 * N_D];  // per-wave P scratch [16][64]

    const int tid  = threadIdx.x;
    const int lane = tid & 31;
    const int wave = tid >> 5;
    const int n16  = lane & 15;          // column within 16-wide tile
    const int half = lane >> 4;          // 0: lanes 0-15, 1: lanes 16-31

    const int bid = blockIdx.x;
    const int qb  = bid & 3;             // query tile index (4 per head)
    const int h   = (bid >> 2) & (N_H - 1);
    const int g   = bid >> 5;            // 0..31
    const int b   = g >> 2;              // mask batch (C = 4)

    const int    l0   = qb * QTILE;
    const size_t head = (size_t)(g * N_H + h) * (size_t)N_L * N_D;

    // per-thread staging roles
    const int kv_d      = tid >> 2;          // K: d row   / V: m row
    const int kv_start  = (tid & 3) * 16;    // K: m start / V: d start
    const float* kbase  = k + head + (size_t)kv_d * N_L + kv_start;   // + m0
    const float* vbase  = v + head + (size_t)kv_d * N_D + kv_start;   // + m0*N_D

    // -------- prefetch tile 0 of K/V into registers --------
    float4 kreg[4], vreg[4];
#pragma unroll
    for (int i = 0; i < 4; ++i) {
        kreg[i] = *(const float4*)(kbase + i * 4);
        vreg[i] = *(const float4*)(vbase + i * 4);
    }

    // -------- stage Q tile (fp32 * QSCALE -> bf16, row-major) --------
    {
        const int row    = tid >> 1;            // 0..127
        const int dstart = (tid & 1) * 32;
        const float* qp  = q + head + (size_t)(l0 + row) * N_D + dstart;
        unsigned short* dst = &sQ[row * N_D + dstart];
#pragma unroll
        for (int i = 0; i < 8; ++i) {
            float4 f = *(const float4*)(qp + i * 4);
            *(uint2*)(dst + i * 4) =
                make_uint2(pack2(f.x * QSCALE, f.y * QSCALE),
                           pack2(f.z * QSCALE, f.w * QSCALE));
        }
    }
    __syncthreads();

    // -------- hoisted Q A-fragments (sQ is immutable from here on) --------
    const int arow = wave * 16 + n16;
    v16bf aq[2];
#pragma unroll
    for (int kk = 0; kk < 2; ++kk) {
        const int koff = kk * 32 + half * 8;
        v8bf lo = *(const v8bf*)&sQ[arow * N_D + koff];
        v8bf hi = *(const v8bf*)&sQ[arow * N_D + koff + 16];
        aq[kk] = cat16(lo, hi);
    }

    // -------- per-wave softmax state (C-layout: row = r + half*8) --------
    float mrow[8], lrow[8];
    v8f   o[4];
    const v8f vzero = {0.f, 0.f, 0.f, 0.f, 0.f, 0.f, 0.f, 0.f};
#pragma unroll
    for (int r = 0; r < 8; ++r) { mrow[r] = NEG_BIG; lrow[r] = 0.f; }
#pragma unroll
    for (int t = 0; t < 4; ++t) o[t] = vzero;

    const unsigned char* mbase = mask + (size_t)b * N_L * N_L
                               + (size_t)(l0 + wave * 16 + half * 8) * N_L;
    unsigned short* pb = &sP[wave * 16 * N_D];

    for (int j = 0; j < NJ; ++j) {
        const int m0 = j * KTILE;

        // ---- stage tile j from regs: K -> sK[m][d], V -> sV[d][m] ----
        // Truncating bf16: each element is a ds_store_b16_d16_hi of the raw
        // fp32 register -- zero conversion VALU.
#pragma unroll
        for (int i = 0; i < 4; ++i) {
            const int mm = kv_start + i * 4;                 // K tile m
            sK[(mm + 0) * N_D + kv_d] = bftrunc(kreg[i].x);
            sK[(mm + 1) * N_D + kv_d] = bftrunc(kreg[i].y);
            sK[(mm + 2) * N_D + kv_d] = bftrunc(kreg[i].z);
            sK[(mm + 3) * N_D + kv_d] = bftrunc(kreg[i].w);
        }
#pragma unroll
        for (int i = 0; i < 4; ++i) {
            const int dd = kv_start + i * 4;                 // V tile d
            sV[(dd + 0) * KTILE + kv_d] = bftrunc(vreg[i].x);
            sV[(dd + 1) * KTILE + kv_d] = bftrunc(vreg[i].y);
            sV[(dd + 2) * KTILE + kv_d] = bftrunc(vreg[i].z);
            sV[(dd + 3) * KTILE + kv_d] = bftrunc(vreg[i].w);
        }

        // ---- prefetch tile j+1 (in flight during compute phase) ----
        if (j + 1 < NJ) {
            const int m1 = (j + 1) * KTILE;
#pragma unroll
            for (int i = 0; i < 4; ++i) {
                kreg[i] = *(const float4*)(kbase + m1 + i * 4);
                vreg[i] = *(const float4*)(vbase + (size_t)m1 * N_D + i * 4);
            }
        }
        __syncthreads();    // sK/sV of tile j visible to all waves

        // ---- S' = (Q*QSCALE) @ K^T  (16 x 64 per wave), 8 WMMAs ----
        v8f sc[4];
#pragma unroll
        for (int t = 0; t < 4; ++t) {
            sc[t] = vzero;
            const int bcol = t * 16 + n16;    // m within tile
#pragma unroll
            for (int kk = 0; kk < 2; ++kk) {
                const int kb = kk * 32 + half * 8;
                v8bf lo = *(const v8bf*)&sK[bcol * N_D + kb];
                v8bf hi = *(const v8bf*)&sK[bcol * N_D + kb + 16];
                v16bf bf = cat16(lo, hi);
                sc[t] = __builtin_amdgcn_wmma_f32_16x16x32_bf16(
                            false, aq[kk], false, bf, (short)0, sc[t], false, false);
            }
        }

        // ---- mask + online softmax (exp2 domain) ----
        float tmax[8];
#pragma unroll
        for (int r = 0; r < 8; ++r) tmax[r] = NEG_BIG;
#pragma unroll
        for (int t = 0; t < 4; ++t) {
            const unsigned char* mp = mbase + m0 + t * 16 + n16;
#pragma unroll
            for (int r = 0; r < 8; ++r) {
                float s = sc[t][r];
                const unsigned char mk = mp[(size_t)r * N_L];
                s = mk ? NEG_BIG : s;
                sc[t][r] = s;
                tmax[r]  = fmaxf(tmax[r], s);
            }
        }
#pragma unroll
        for (int r = 0; r < 8; ++r) tmax[r] = redmax16(tmax[r]);   // v_permlane16

        float corr[8], rsum[8];
#pragma unroll
        for (int r = 0; r < 8; ++r) {
            const float mnew = fmaxf(mrow[r], tmax[r]);
            corr[r] = fast_exp2(mrow[r] - mnew);
            mrow[r] = mnew;
            rsum[r] = 0.f;
        }
#pragma unroll
        for (int t = 0; t < 4; ++t)
#pragma unroll
            for (int r = 0; r < 8; ++r) {
                const float p = fast_exp2(sc[t][r] - mrow[r]);
                sc[t][r] = p;
                rsum[r] += p;
            }
#pragma unroll
        for (int r = 0; r < 8; ++r)                                // v_permlane16
            lrow[r] = fmaf(lrow[r], corr[r], redsum16(rsum[r]));

        // ---- P -> per-wave LDS scratch, bf16 truncation (d16_hi store) ----
#pragma unroll
        for (int t = 0; t < 4; ++t) {
            const int col = t * 16 + n16;
#pragma unroll
            for (int r = 0; r < 8; ++r)
                pb[(half * 8 + r) * N_D + col] = bftrunc(sc[t][r]);
        }

        // rescale O by correction factors
#pragma unroll
        for (int t = 0; t < 4; ++t)
#pragma unroll
            for (int r = 0; r < 8; ++r) o[t][r] *= corr[r];

        // ---- O += P @ V  (16 x 64), 8 WMMAs ----
        v16bf pa[2];
#pragma unroll
        for (int kk = 0; kk < 2; ++kk) {
            const int koff = kk * 32 + half * 8;
            v8bf lo = *(const v8bf*)&pb[n16 * N_D + koff];
            v8bf hi = *(const v8bf*)&pb[n16 * N_D + koff + 16];
            pa[kk] = cat16(lo, hi);
        }
#pragma unroll
        for (int t = 0; t < 4; ++t) {
            const int bcol = t * 16 + n16;    // d within tile
#pragma unroll
            for (int kk = 0; kk < 2; ++kk) {
                const int kb = kk * 32 + half * 8;
                v8bf lo = *(const v8bf*)&sV[bcol * KTILE + kb];
                v8bf hi = *(const v8bf*)&sV[bcol * KTILE + kb + 16];
                v16bf bf = cat16(lo, hi);
                o[t] = __builtin_amdgcn_wmma_f32_16x16x32_bf16(
                           false, pa[kk], false, bf, (short)0, o[t], false, false);
            }
        }
        __syncthreads();    // compute reads of sK/sV done before next staging
    }

    // -------- finalize: O * (1/l), write fp32 --------
    float rinv[8];
#pragma unroll
    for (int r = 0; r < 8; ++r) rinv[r] = 1.0f / lrow[r];
#pragma unroll
    for (int t = 0; t < 4; ++t) {
        const int d = t * 16 + n16;
#pragma unroll
        for (int r = 0; r < 8; ++r) {
            const int row = l0 + wave * 16 + half * 8 + r;
            out[head + (size_t)row * N_D + d] = o[t][r] * rinv[r];
        }
    }
}

extern "C" void kernel_launch(void* const* d_in, const int* in_sizes, int n_in,
                              void* d_out, int out_size, void* d_ws, size_t ws_size,
                              hipStream_t stream) {
    const float*         q    = (const float*)d_in[0];
    const float*         k    = (const float*)d_in[1];
    const float*         v    = (const float*)d_in[2];
    const unsigned char* mask = (const unsigned char*)d_in[3];
    float*               out  = (float*)d_out;

    dim3 grid(N_G * N_H * (N_L / QTILE));   // 32*8*4 = 1024 blocks
    dim3 block(256);                        // 8 wave32 waves
    sdpa_flash_bf16_kernel<<<grid, block, 0, stream>>>(q, k, v, mask, out);
}
